// RNN_GRU_8340826489350
// MI455X (gfx1250) — matrix operand order repaired and checked by reading, not compiled, not run on hardware
//
#include <hip/hip_runtime.h>
#include <hip/hip_bf16.h>
#include <math.h>
#include <stdint.h>

// Problem dims
#define B_   64
#define T_   512
#define IN_  128
#define H_   256
#define L_   3
#define OUT_ 50
#define G_   768   // 3*H

typedef __bf16 bf16_t;
typedef __attribute__((ext_vector_type(16))) __bf16 v16bf;
typedef __attribute__((ext_vector_type(8)))  __bf16 v8bf;
typedef __attribute__((ext_vector_type(8)))  float  v8f;

#define AS1 __attribute__((address_space(1)))
#define AS3 __attribute__((address_space(3)))

// gcc-style vector type matching the builtin's expected parameter type
typedef int v4i_vs __attribute__((__vector_size__(4 * sizeof(int))));

#if defined(__has_builtin)
#if __has_builtin(__builtin_amdgcn_global_load_async_to_lds_b128)
#define HAVE_ASYNC_BUILTIN 1
#endif
#if __has_builtin(__builtin_amdgcn_s_wait_asynccnt)
#define HAVE_WAIT_ASYNC_BUILTIN 1
#endif
#endif

// Async global->LDS 16-byte copy (GLOBAL_LOAD_ASYNC_TO_LDS_B128, ASYNCcnt).
__device__ __forceinline__ void async_load_b128(const void* g, void* l) {
#if defined(HAVE_ASYNC_BUILTIN)
  __builtin_amdgcn_global_load_async_to_lds_b128(
      (AS1 v4i_vs*)(uintptr_t)g,
      (AS3 v4i_vs*)(uint32_t)(uintptr_t)l, 0, 0);
#else
  asm volatile("global_load_async_to_lds_b128 %0, %1, off"
               :: "v"((uint32_t)(uintptr_t)l),
                  "v"((uint64_t)(uintptr_t)g)
               : "memory");
#endif
}

__device__ __forceinline__ void wait_async0() {
#if defined(HAVE_WAIT_ASYNC_BUILTIN)
  __builtin_amdgcn_s_wait_asynccnt(0);
#else
  asm volatile("s_wait_asynccnt 0" ::: "memory");
#endif
}

// ---------------------------------------------------------------------------
// WMMA fragment loaders (wave32, 16x16x32 bf16; layouts per CDNA5 ISA 7.12.2)
// ---------------------------------------------------------------------------
__device__ __forceinline__ v16bf load_frag_a(const bf16_t* base, int ld) {
  const int lane = threadIdx.x & 31;
  const int row  = lane & 15;
  const int k0   = (lane < 16) ? 0 : 8;
  const bf16_t* p = base + (size_t)row * ld + k0;
  v8bf lo = *reinterpret_cast<const v8bf*>(p);
  v8bf hi = *reinterpret_cast<const v8bf*>(p + 16);
  v16bf f;
#pragma unroll
  for (int i = 0; i < 8; ++i) { f[i] = lo[i]; f[i + 8] = hi[i]; }
  return f;
}

__device__ __forceinline__ v16bf load_frag_b(const bf16_t* wbase, int ld) {
  const int lane = threadIdx.x & 31;
  const int n    = lane & 15;
  const int k0   = (lane < 16) ? 0 : 8;
  const bf16_t* p = wbase + (size_t)n * ld + k0;
  v8bf lo = *reinterpret_cast<const v8bf*>(p);
  v8bf hi = *reinterpret_cast<const v8bf*>(p + 16);
  v16bf f;
#pragma unroll
  for (int i = 0; i < 8; ++i) { f[i] = lo[i]; f[i + 8] = hi[i]; }
  return f;
}

// ---------------------------------------------------------------------------
// f32 -> bf16 elementwise cast
// ---------------------------------------------------------------------------
__global__ void cast_f32_bf16(const float* __restrict__ src,
                              bf16_t* __restrict__ dst, long n) {
  long i = (long)blockIdx.x * blockDim.x + threadIdx.x;
  const long stride = (long)gridDim.x * blockDim.x;
  for (; i < n; i += stride) dst[i] = (bf16_t)src[i];
}

// ---------------------------------------------------------------------------
// Input projection GEMM: xg[dir][m][g] = sum_k A[m,k]*W[dir][g,k] + bih[dir][g]
// Block = 256 thr (8 waves); workgroup tile 128(M) x 64(N).
// W chunks double-buffered in LDS via async global->LDS loads.
// ---------------------------------------------------------------------------
__global__ void __launch_bounds__(256)
xproj_gemm(const bf16_t* __restrict__ A, const bf16_t* __restrict__ Wg,
           const float* __restrict__ bias, float* __restrict__ out, int D) {
  const int dir  = blockIdx.z;
  const int tid  = threadIdx.x;
  const int wave = tid >> 5;
  __shared__ bf16_t sW[2][64 * 32];

  const bf16_t* Wd    = Wg + (size_t)dir * G_ * D;
  const size_t  mbase = (size_t)blockIdx.x * 128 + (size_t)wave * 16;
  const int     gbase = blockIdx.y * 64;

  // each thread owns one 16B chunk of the 64x32 bf16 stage
  const int sr = tid >> 2, sc = (tid & 3) * 8;

  v8f acc[4] = {};
  const int kchunks = D >> 5;

  // prologue: stage chunk 0 into buffer 0
  async_load_b128(&Wd[(size_t)(gbase + sr) * D + 0 * 32 + sc], &sW[0][sr * 32 + sc]);

  int buf = 0;
  for (int kc = 0; kc < kchunks; ++kc) {
    wait_async0();
    __syncthreads();
    if (kc + 1 < kchunks)
      async_load_b128(&Wd[(size_t)(gbase + sr) * D + (kc + 1) * 32 + sc],
                      &sW[buf ^ 1][sr * 32 + sc]);
    const v16bf a = load_frag_a(A + mbase * D + kc * 32, D);
#pragma unroll
    for (int nt = 0; nt < 4; ++nt) {
      const v16bf b = load_frag_b(&sW[buf][nt * 16 * 32], 32);
      acc[nt] = __builtin_amdgcn_wmma_f32_16x16x32_bf16(
          false, a, false, b, (short)0, acc[nt], false, false);
    }
    buf ^= 1;
  }

  // write out + bias (C layout: VGPR r, lanes0-15 -> M=r, lanes16-31 -> M=r+8)
  const int lane = tid & 31;
  const int n    = lane & 15;
  const int mb   = (lane < 16) ? 0 : 8;
  float* outd = out + (size_t)dir * (B_ * T_) * G_;
#pragma unroll
  for (int nt = 0; nt < 4; ++nt) {
    const int   g  = gbase + nt * 16 + n;
    const float bv = bias[dir * G_ + g];
#pragma unroll
    for (int r = 0; r < 8; ++r)
      outd[(mbase + mb + r) * (size_t)G_ + g] = acc[nt][r] + bv;
  }
}

// ---------------------------------------------------------------------------
// Recurrent scan: one workgroup per direction, 512 thr (16 waves).
// Per step: hg(64x768) = h_bf16(64x256) @ Whh^T via WMMA into LDS,
// then gate math on per-thread f32 h registers; h fed back as bf16 in LDS.
// Dynamic LDS: hg f32 (192KB) + h bf16 (32KB) = 224KB (fits 320KB WGP LDS).
// ---------------------------------------------------------------------------
__global__ void __launch_bounds__(512)
gru_scan(const float* __restrict__ xg, const bf16_t* __restrict__ Whh,
         const float* __restrict__ bhh, float* __restrict__ hout) {
  const int dir = blockIdx.x;
  extern __shared__ char smem[];
  float*  hg  = reinterpret_cast<float*>(smem);                            // 64*768
  bf16_t* hbf = reinterpret_cast<bf16_t*>(smem + 64 * G_ * sizeof(float)); // 64*256

  const int tid = threadIdx.x;
  for (int i = tid; i < 64 * H_; i += 512) hbf[i] = (bf16_t)0.0f;
  float hreg[32];
#pragma unroll
  for (int i = 0; i < 32; ++i) hreg[i] = 0.0f;
  __syncthreads();

  const bf16_t* Wd  = Whh + (size_t)dir * G_ * H_;
  const float*  bd  = bhh + dir * G_;
  const float*  xgd = xg + (size_t)dir * (B_ * T_) * G_;

  const int wave = tid >> 5;
  const int mt   = wave & 3;    // M subtile (16 rows)
  const int ntb  = wave >> 2;   // base N subtile, stride 4 over 48
  const int lane = tid & 31;
  const int ln   = lane & 15;
  const int lmb  = (lane < 16) ? 0 : 8;
  const int ub   = tid >> 3;        // batch row this thread updates
  const int uj   = (tid & 7) * 32;  // hidden-col base this thread updates

  for (int tt = 0; tt < T_; ++tt) {
    const int t = dir ? (T_ - 1 - tt) : tt;

    // prefetch this step's gate-preactivation rows while WMMAs run
    const float* xgt = xgd + ((size_t)ub * T_ + t) * G_;
    __builtin_prefetch(xgt + uj, 0, 1);
    __builtin_prefetch(xgt + H_ + uj, 0, 1);
    __builtin_prefetch(xgt + 2 * H_ + uj, 0, 1);

    // Preload this wave's A fragments (h tile 16x256) from LDS once.
    v16bf afr[8];
#pragma unroll
    for (int kc = 0; kc < 8; ++kc)
      afr[kc] = load_frag_a(hbf + mt * 16 * H_ + kc * 32, H_);

#pragma unroll 1
    for (int i = 0; i < 12; ++i) {
      const int nt = ntb + 4 * i;
      const bf16_t* wb = Wd + (size_t)(nt * 16) * H_;
      if (i + 1 < 12)
        __builtin_prefetch(Wd + (size_t)((ntb + 4 * (i + 1)) * 16) * H_, 0, 1);
      v8f acc = {};
#pragma unroll
      for (int kc = 0; kc < 8; ++kc) {
        const v16bf b = load_frag_b(wb + kc * 32, H_);
        acc = __builtin_amdgcn_wmma_f32_16x16x32_bf16(
            false, afr[kc], false, b, (short)0, acc, false, false);
      }
      const int col  = nt * 16 + ln;
      const int rowb = mt * 16 + lmb;
#pragma unroll
      for (int r = 0; r < 8; ++r) hg[(rowb + r) * G_ + col] = acc[r];
    }
    __syncthreads();

    // Gate update: r,z,n -> new h
#pragma unroll 4
    for (int i = 0; i < 32; ++i) {
      const int j = uj + i;
      const float xr = xgt[j], xz = xgt[H_ + j], xn = xgt[2 * H_ + j];
      const float hr = hg[ub * G_ + j]          + bd[j];
      const float hz = hg[ub * G_ + H_ + j]     + bd[H_ + j];
      const float hn = hg[ub * G_ + 2 * H_ + j] + bd[2 * H_ + j];
      const float rg = 1.0f / (1.0f + __expf(-(xr + hr)));
      const float zg = 1.0f / (1.0f + __expf(-(xz + hz)));
      const float ng = tanhf(xn + rg * hn);
      const float h  = (1.0f - zg) * ng + zg * hreg[i];
      hreg[i] = h;
      hbf[ub * H_ + j] = (bf16_t)h;
      hout[((size_t)ub * T_ + t) * (2 * H_) + dir * H_ + j] = h;
    }
    __syncthreads();
  }
}

// ---------------------------------------------------------------------------
// MLP head on h[:, T-1, :]  (tiny: ~12 MFLOP, scalar f32)
// ---------------------------------------------------------------------------
__global__ void __launch_bounds__(256)
head_mlp(const float* __restrict__ hfin,
         const float* __restrict__ W1, const float* __restrict__ b1,
         const float* __restrict__ W2, const float* __restrict__ b2,
         const float* __restrict__ W3, const float* __restrict__ b3,
         const float* __restrict__ W4, const float* __restrict__ b4,
         float* __restrict__ out) {
  extern __shared__ float sm[];
  float* y1 = sm;                      // 64*128
  float* y2 = sm + 64 * 128;           // 64*64
  float* y3 = sm + 64 * 128 + 64 * 64; // 64*256
  const int tid = threadIdx.x;

  for (int idx = tid; idx < 64 * 128; idx += 256) {
    const int b = idx >> 7, o = idx & 127;
    const float* xr = hfin + ((size_t)b * T_ + (T_ - 1)) * (2 * H_);
    const float* wr = W1 + (size_t)o * (2 * H_);
    float s = b1[o];
    for (int k = 0; k < 2 * H_; ++k) s += xr[k] * wr[k];
    y1[idx] = fmaxf(s, 0.0f);
  }
  __syncthreads();
  for (int idx = tid; idx < 64 * 64; idx += 256) {
    const int b = idx >> 6, o = idx & 63;
    const float* wr = W2 + o * 128;
    float s = b2[o];
    for (int k = 0; k < 128; ++k) s += y1[b * 128 + k] * wr[k];
    y2[idx] = fmaxf(s, 0.0f);
  }
  __syncthreads();
  for (int idx = tid; idx < 64 * 256; idx += 256) {
    const int b = idx >> 8, o = idx & 255;
    const float* wr = W3 + o * 64;
    float s = b3[o];
    for (int k = 0; k < 64; ++k) s += y2[b * 64 + k] * wr[k];
    y3[idx] = fmaxf(s, 0.0f);
  }
  __syncthreads();
  for (int idx = tid; idx < 64 * OUT_; idx += 256) {
    const int b = idx / OUT_, o = idx % OUT_;
    const float* wr = W4 + o * H_;
    float s = b4[o];
    for (int k = 0; k < H_; ++k) s += y3[b * H_ + k] * wr[k];
    out[idx] = s;
  }
}

// ---------------------------------------------------------------------------
extern "C" void kernel_launch(void* const* d_in, const int* in_sizes, int n_in,
                              void* d_out, int out_size, void* d_ws, size_t ws_size,
                              hipStream_t stream) {
  const float* x    = (const float*)d_in[0];
  const float* Wih0 = (const float*)d_in[1];
  const float* Whh0 = (const float*)d_in[2];
  const float* bih0 = (const float*)d_in[3];
  const float* bhh0 = (const float*)d_in[4];
  const float* WihL = (const float*)d_in[5];
  const float* WhhL = (const float*)d_in[6];
  const float* bihL = (const float*)d_in[7];
  const float* bhhL = (const float*)d_in[8];
  const float* W1 = (const float*)d_in[9];  const float* b1 = (const float*)d_in[10];
  const float* W2 = (const float*)d_in[11]; const float* b2 = (const float*)d_in[12];
  const float* W3 = (const float*)d_in[13]; const float* b3 = (const float*)d_in[14];
  const float* W4 = (const float*)d_in[15]; const float* b4 = (const float*)d_in[16];
  float* out = (float*)d_out;

  char* ws = (char*)d_ws;
  size_t off = 0;
  auto alloc = [&](size_t bytes) -> void* {
    off = (off + 255) & ~(size_t)255;
    void* p = ws + off;
    off += bytes;
    return p;
  };
  bf16_t* xin_bf = (bf16_t*)alloc((size_t)B_ * T_ * 512 * sizeof(bf16_t));
  bf16_t* wih_bf = (bf16_t*)alloc((size_t)2 * G_ * 512 * sizeof(bf16_t));
  bf16_t* whh_bf = (bf16_t*)alloc((size_t)2 * G_ * H_ * sizeof(bf16_t));
  float*  xgbuf  = (float*)alloc((size_t)2 * B_ * T_ * G_ * sizeof(float));
  float*  hbuf0  = (float*)alloc((size_t)B_ * T_ * 2 * H_ * sizeof(float));
  float*  hbuf1  = (float*)alloc((size_t)B_ * T_ * 2 * H_ * sizeof(float));

  const float* lin = x;
  float* louts[3] = {hbuf0, hbuf1, hbuf0};
  for (int l = 0; l < L_; ++l) {
    const int D = (l == 0) ? IN_ : 2 * H_;
    const float* wi = (l == 0) ? Wih0 : WihL + (size_t)(l - 1) * 2 * G_ * 2 * H_;
    const float* wh = (l == 0) ? Whh0 : WhhL + (size_t)(l - 1) * 2 * G_ * H_;
    const float* bi = (l == 0) ? bih0 : bihL + (size_t)(l - 1) * 2 * G_;
    const float* bh = (l == 0) ? bhh0 : bhhL + (size_t)(l - 1) * 2 * G_;

    cast_f32_bf16<<<512, 256, 0, stream>>>(lin, xin_bf, (long)B_ * T_ * D);
    cast_f32_bf16<<<256, 256, 0, stream>>>(wi, wih_bf, (long)2 * G_ * D);
    cast_f32_bf16<<<256, 256, 0, stream>>>(wh, whh_bf, (long)2 * G_ * H_);

    dim3 gdim((B_ * T_) / 128, G_ / 64, 2);
    xproj_gemm<<<gdim, 256, 0, stream>>>(xin_bf, wih_bf, bi, xgbuf, D);

    const size_t scan_lds = (size_t)64 * G_ * sizeof(float) + (size_t)64 * H_ * sizeof(bf16_t);
    gru_scan<<<2, 512, scan_lds, stream>>>(xgbuf, whh_bf, bh, louts[l]);
    lin = louts[l];
  }
  const size_t head_lds = (size_t)(64 * 128 + 64 * 64 + 64 * 256) * sizeof(float);
  head_mlp<<<1, 256, head_lds, stream>>>(louts[2], W1, b1, W2, b2, W3, b3, W4, b4, out);
}
